// HeteroPNANet_8830452760918
// MI455X (gfx1250) — matrix operand order repaired and checked by reading, not compiled, not run on hardware
//
#include <hip/hip_runtime.h>

// =====================================================================
// HeteroPNANet for gfx1250 (MI455X, wave32, WMMA, TDM).
// Dense matmuls -> v_wmma_f32_16x16x32_bf16 (f32 HBM activations,
// bf16 conversion at the LDS stage, f32 accumulate).
// A-tiles DMA'd into LDS by the Tensor Data Mover (TENSORcnt-tracked);
// B-tiles staged with branch-free vectorized loads.
// Edge segment ops -> f32 atomics (add native, min/max via CAS).
// =====================================================================

typedef __attribute__((ext_vector_type(16))) __bf16 v16bf;
typedef __attribute__((ext_vector_type(8)))  float  v8f;
typedef __attribute__((ext_vector_type(4)))  unsigned int u32x4;
typedef __attribute__((ext_vector_type(8)))  int i32x8;
typedef __attribute__((ext_vector_type(4)))  int i32x4;

#if __has_builtin(__builtin_amdgcn_tensor_load_to_lds)
#define HAVE_TDM 1
#else
#define HAVE_TDM 0
#endif

#define GEMM_ACC  1
#define GEMM_BIAS 2

__device__ __forceinline__ __bf16 f2bf(float f) { return (__bf16)f; }

__device__ __forceinline__ void atomicMaxF(float* addr, float val) {
    unsigned int* ai = (unsigned int*)addr;
    unsigned int old = __hip_atomic_load(ai, __ATOMIC_RELAXED, __HIP_MEMORY_SCOPE_AGENT);
    while (true) {
        float f = __uint_as_float(old);
        if (f >= val) break;
        unsigned int assumed = old;
        old = atomicCAS(ai, assumed, __float_as_uint(val));
        if (old == assumed) break;
    }
}
__device__ __forceinline__ void atomicMinF(float* addr, float val) {
    unsigned int* ai = (unsigned int*)addr;
    unsigned int old = __hip_atomic_load(ai, __ATOMIC_RELAXED, __HIP_MEMORY_SCOPE_AGENT);
    while (true) {
        float f = __uint_as_float(old);
        if (f <= val) break;
        unsigned int assumed = old;
        old = atomicCAS(ai, assumed, __float_as_uint(val));
        if (old == assumed) break;
    }
}

// ---------------------------------------------------------------------
// TDM: issue an async 2D tile load (tile_x x tile_y elements of 4B) from
// global (row stride = stride0 elements) into LDS at byte offset lds_off.
// D# layout per CDNA5 ISA ch.8 (group0: count/lds/global/type,
// group1: data_size, tensor dims, tile dims, dim0 stride).
// ---------------------------------------------------------------------
__device__ __forceinline__ void tdm_load_2d_f32(const void* gaddr, unsigned lds_off,
                                                int tile_x, int tile_y,
                                                unsigned tdim0, unsigned tdim1,
                                                unsigned long long stride0) {
#if HAVE_TDM
    unsigned long long ga = (unsigned long long)gaddr;
    u32x4 g0;
    g0[0] = 1u;                                   // count=1, user descriptor
    g0[1] = lds_off;                              // lds_addr (bytes)
    g0[2] = (unsigned)(ga & 0xffffffffu);         // global_addr[31:0]
    g0[3] = (unsigned)((ga >> 32) & 0x01ffffffu)  // global_addr[56:32]
          | (2u << 30);                           // type=2 ("image")
    i32x8 g1;
    g1[0] = (int)(2u << 16);                      // data_size = 2 -> 4 bytes
    g1[1] = (int)((tdim0 & 0xffffu) << 16);       // tensor_dim0[15:0] at bit48
    g1[2] = (int)(((tdim0 >> 16) & 0xffffu) | ((tdim1 & 0xffffu) << 16));
    g1[3] = (int)(((tdim1 >> 16) & 0xffffu) | (((unsigned)tile_x & 0xffffu) << 16));
    g1[4] = (int)((unsigned)tile_y & 0xffffu);    // tile_dim1; tile_dim2 = 0
    g1[5] = (int)(stride0 & 0xffffffffu);         // tensor_dim0_stride[31:0]
    g1[6] = (int)((stride0 >> 32) & 0xffffu);     // stride[47:32]; dim1_stride lo = 0
    g1[7] = 0;
    i32x4 z4 = {0, 0, 0, 0};
#if defined(__clang_major__) && (__clang_major__ >= 23)
    i32x8 z8 = {0, 0, 0, 0, 0, 0, 0, 0};
    __builtin_amdgcn_tensor_load_to_lds(g0, g1, z4, z4, z8, 0);
#else
    __builtin_amdgcn_tensor_load_to_lds(g0, g1, z4, z4, 0);
#endif
#endif
}

// ---------------------------------------------------------------------
// Generic WMMA GEMM: C[M,N] (+)= A[M,K] @ W[K,N] (+ bias)
// Preconditions (hold for every call in this net): M%32==0, K%32==0, N%4==0.
// Block = 256 threads = 8 waves, block tile 32(M) x 64(N); wave grid 2x4,
// each wave computes one 16x16 tile via v_wmma_f32_16x16x32_bf16.
// A tile: TDM async DMA (f32, 32x32) -> LDS; converted to bf16 at
//         fragment build (contiguous float2 + v_cvt_pk_bf16_f32).
// B tile: branch-free float4 loads, f32->bf16, stored transposed [n][k]
//         so fragment K-pairs are contiguous (merge to ds_load_2addr_b64).
//
// CDNA5 wave32 fragment layouts (ISA 7.12.2):
//  A 16x32 bf16: lane = kh*16+m; VGPR v holds K = base,base+1,
//                base = (v<4 ? 2v : 8+2v) + 8*kh  (two contiguous 8-K runs)
//  B 32x16 bf16: lane = kh*16+n; VGPR v holds K = base,base+1 (same as A)
//  C/D 16x16 f32: lanes 0-15: N=lane, VGPR r -> M=r; lanes 16-31: M=r+8
// ---------------------------------------------------------------------
__global__ __launch_bounds__(256)
void wmma_gemm_kernel(const float* __restrict__ A, int lda,
                      const float* __restrict__ W, int ldw,
                      const float* __restrict__ bias,
                      float* __restrict__ C, int ldc,
                      int M, int N, int K, int flags)
{
    __shared__ float  lAf[32][32];   // f32 A tile (TDM target, packed row-major)
    __shared__ __bf16 lB[64][36];    // bf16 B tile, transposed [n][k], padded

    const int tid  = threadIdx.x;
    const int lane = tid & 31;
    const int wave = tid >> 5;       // 0..7
    const int wm   = wave >> 2;      // 0..1
    const int wn   = wave & 3;       // 0..3

    const int rowBase = blockIdx.x * 32;
    const int colBase = blockIdx.y * 64;

    v8f acc = {0.f, 0.f, 0.f, 0.f, 0.f, 0.f, 0.f, 0.f};

    for (int k0 = 0; k0 < K; k0 += 32) {
#if HAVE_TDM
        if (wave == 0) {   // async DMA of the 32x32 f32 A tile into LDS
            tdm_load_2d_f32(&A[(size_t)rowBase * lda + k0],
                            (unsigned)(size_t)&lAf[0][0],
                            32, 32, (unsigned)K, (unsigned)M,
                            (unsigned long long)lda);
        }
#endif
        if (k0 + 32 < K) {  // prefetch next W panel while this tile computes
            __builtin_prefetch(&W[(size_t)(k0 + 32 + (tid >> 4)) * ldw + colBase], 0, 3);
        }
        // ---- stage B: 32k x 64n -> lB[n][k]; branch-free (N%4==0 so the
        //      float4 either lies fully inside or fully outside [0,N))
        {
            const int n4 = (tid & 15) * 4;
            const int gn = colBase + n4;
            const int gnc = (gn + 3 < N) ? gn : (N >= 4 ? N - 4 : 0);
            const bool ok = (gn < N);
            #pragma unroll
            for (int i = 0; i < 2; ++i) {
                const int k = (tid >> 4) + i * 16;
                const float4 w4 = *(const float4*)&W[(size_t)(k0 + k) * ldw + gnc];
                lB[n4 + 0][k] = f2bf(ok ? w4.x : 0.f);
                lB[n4 + 1][k] = f2bf(ok ? w4.y : 0.f);
                lB[n4 + 2][k] = f2bf(ok ? w4.z : 0.f);
                lB[n4 + 3][k] = f2bf(ok ? w4.w : 0.f);
            }
        }
#if HAVE_TDM
        if (wave == 0) {
#if __has_builtin(__builtin_amdgcn_s_wait_tensorcnt)
            __builtin_amdgcn_s_wait_tensorcnt(0);
#endif
        }
#else
        // fallback: cooperative b128 copy of the A tile (M%32==0, K%32==0)
        {
            const int r  = tid >> 3;
            const int kq = (tid & 7) * 4;
            *(float4*)&lAf[r][kq] = *(const float4*)&A[(size_t)(rowBase + r) * lda + k0 + kq];
        }
#endif
        __syncthreads();

        // ---- build fragments
        const int am = lane & 15;
        const int kh = lane >> 4;
        v16bf afrag, bfrag;
        #pragma unroll
        for (int vv = 0; vv < 8; ++vv) {
            const int kb = ((vv < 4) ? (2 * vv) : (8 + 2 * vv)) + 8 * kh;
            const float2 p = *(const float2*)&lAf[wm * 16 + am][kb];
            afrag[2 * vv]     = f2bf(p.x);   // pairs -> v_cvt_pk_bf16_f32
            afrag[2 * vv + 1] = f2bf(p.y);
        }
        const int bn  = lane & 15;
        const int bkb = (lane >> 4) * 8;
        #pragma unroll
        for (int vv = 0; vv < 8; ++vv) {
            const int kk = ((vv < 4) ? (2 * vv) : (8 + 2 * vv)) + bkb;
            bfrag[2 * vv]     = lB[wn * 16 + bn][kk];
            bfrag[2 * vv + 1] = lB[wn * 16 + bn][kk + 1];
        }
        acc = __builtin_amdgcn_wmma_f32_16x16x32_bf16(false, afrag, false, bfrag,
                                                      (short)0, acc, false, false);
        __syncthreads();
    }

    // ---- store C (f32 16x16 layout); rows always in range (M%32==0)
    const int cn   = lane & 15;
    const int rAdd = (lane >> 4) * 8;
    #pragma unroll
    for (int r = 0; r < 8; ++r) {
        const int row = rowBase + wm * 16 + rAdd + r;
        const int col = colBase + wn * 16 + cn;
        if (col < N) {
            float v = acc[r];
            if (flags & GEMM_BIAS) v += bias[col];
            size_t o = (size_t)row * ldc + col;
            if (flags & GEMM_ACC) v += C[o];
            C[o] = v;
        }
    }
}

// ---------------------------------------------------------------------
// Elementwise / edge kernels
// ---------------------------------------------------------------------
__global__ void fill_kernel(float* __restrict__ p, float v, long long n) {
    long long i = (long long)blockIdx.x * blockDim.x + threadIdx.x;
    if (i < n) p[i] = v;
}

__global__ void count_kernel(const int* __restrict__ dst, int E, float* __restrict__ cnt) {
    long long i = (long long)blockIdx.x * blockDim.x + threadIdx.x;
    if (i < E) atomicAdd(&cnt[dst[i]], 1.0f);
}

__global__ void degscale_kernel(const float* __restrict__ cnt, const float* __restrict__ avg,
                                float* __restrict__ amp, float* __restrict__ att, int N) {
    int i = blockIdx.x * blockDim.x + threadIdx.x;
    if (i >= N) return;
    float dl = logf(fmaxf(cnt[i], 1.f) + 1.f);
    float a  = avg[0];
    amp[i] = dl / a;
    att[i] = a / dl;
}

// aggr row layout: [mean(s1) | max | min | std(s2)] chunks of F, row stride 4F
__global__ void pna_aggr_init_kernel(float* __restrict__ aggr, int N, int F) {
    long long i = (long long)blockIdx.x * blockDim.x + threadIdx.x;
    long long total = (long long)N * 4 * F;
    if (i >= total) return;
    int chunk = (int)((i % (4 * F)) / F);
    float v = 0.f;
    if (chunk == 1) v = -3.0e38f;
    else if (chunk == 2) v = 3.0e38f;
    aggr[i] = v;
}

__global__ void pna_edge_kernel(const int* __restrict__ src, const int* __restrict__ dst,
                                const float* __restrict__ pre_i, const float* __restrict__ pre_j,
                                const float* __restrict__ bpre,
                                float* __restrict__ aggr, int E, int F) {
    long long i = (long long)blockIdx.x * blockDim.x + threadIdx.x;
    long long total = (long long)E * F;
    if (i >= total) return;
    int e = (int)(i / F), f = (int)(i % F);
    int s = src[e], d = dst[e];
    float m = pre_i[(size_t)d * F + f] + pre_j[(size_t)s * F + f] + bpre[f];
    float* row = aggr + (size_t)d * (4 * F);
    atomicAdd(&row[f], m);
    atomicMaxF(&row[F + f], m);
    atomicMinF(&row[2 * F + f], m);
    atomicAdd(&row[3 * F + f], m * m);
}

__global__ void pna_finalize_kernel(float* __restrict__ aggr, const float* __restrict__ cnt,
                                    int N, int F) {
    long long i = (long long)blockIdx.x * blockDim.x + threadIdx.x;
    long long total = (long long)N * F;
    if (i >= total) return;
    int n = (int)(i / F), f = (int)(i % F);
    float c  = cnt[n];
    float cc = fmaxf(c, 1.f);
    float* row = aggr + (size_t)n * 4 * F;
    float s1 = row[f], s2 = row[3 * F + f];
    float mean = s1 / cc;
    float var  = fmaxf(s2 / cc - mean * mean, 0.f);
    row[f] = mean;
    row[3 * F + f] = sqrtf(var + 1e-5f);
    if (c <= 0.f) { row[F + f] = 0.f; row[2 * F + f] = 0.f; }  // isfinite->0 convention
}

// out += amp[n]*z2 + att[n]*z3 + b_post[f]
__global__ void pna_combine_kernel(float* __restrict__ out, const float* __restrict__ z2,
                                   const float* __restrict__ z3, const float* __restrict__ amp,
                                   const float* __restrict__ att, const float* __restrict__ bpost,
                                   int N, int F) {
    long long i = (long long)blockIdx.x * blockDim.x + threadIdx.x;
    long long total = (long long)N * F;
    if (i >= total) return;
    int n = (int)(i / F), f = (int)(i % F);
    out[i] += amp[n] * z2[i] + att[n] * z3[i] + bpost[f];
}

__global__ __launch_bounds__(256)
void layernorm_kernel(const float* __restrict__ x, const float* __restrict__ g,
                      const float* __restrict__ b, const float* __restrict__ resid,
                      float* __restrict__ out, int F, int relu)
{
    int row = blockIdx.x;
    const float* xr = x + (size_t)row * F;
    float s = 0.f, ss = 0.f;
    for (int f = threadIdx.x; f < F; f += blockDim.x) { float v = xr[f]; s += v; ss += v * v; }
    #pragma unroll
    for (int o = 16; o > 0; o >>= 1) { s += __shfl_xor(s, o, 32); ss += __shfl_xor(ss, o, 32); }
    __shared__ float shs[8], shss[8];
    int lane = threadIdx.x & 31, wv = threadIdx.x >> 5;
    if (lane == 0) { shs[wv] = s; shss[wv] = ss; }
    __syncthreads();
    s = 0.f; ss = 0.f;
    int nw = (blockDim.x + 31) >> 5;
    for (int w = 0; w < nw; ++w) { s += shs[w]; ss += shss[w]; }
    float mean = s / F;
    float rstd = rsqrtf(ss / F - mean * mean + 1e-5f);
    for (int f = threadIdx.x; f < F; f += blockDim.x) {
        float v = (xr[f] - mean) * rstd * g[f] + b[f];
        if (relu) v = fmaxf(v, 0.f);
        if (resid) v += resid[(size_t)row * F + f];
        out[(size_t)row * F + f] = v;
    }
}

// HGT attention passes (H = 4 heads)
__global__ void hgt_logit_kernel(const int* __restrict__ src, const int* __restrict__ dst,
                                 const float* __restrict__ q, const float* __restrict__ krel,
                                 const float* __restrict__ prior,
                                 float* __restrict__ lg, float* __restrict__ m,
                                 int E, int fo, int D, int off_d, int relrow) {
    long long i = (long long)blockIdx.x * blockDim.x + threadIdx.x;
    long long total = (long long)E * 4;
    if (i >= total) return;
    int e = (int)(i >> 2), h = (int)(i & 3);
    int d = dst[e], s = src[e];
    const float* qp = q    + (size_t)(off_d + d)   * fo + h * D;
    const float* kp = krel + (size_t)(relrow + s)  * fo + h * D;
    float dot = 0.f;
    for (int j = 0; j < D; ++j) dot += qp[j] * kp[j];
    float l = dot * prior[h] * rsqrtf((float)D);
    lg[(size_t)e * 4 + h] = l;
    atomicMaxF(&m[(size_t)(off_d + d) * 4 + h], l);
}

__global__ void hgt_exp_kernel(const int* __restrict__ dst, float* __restrict__ lg,
                               const float* __restrict__ m, float* __restrict__ z,
                               int E, int off_d) {
    long long i = (long long)blockIdx.x * blockDim.x + threadIdx.x;
    long long total = (long long)E * 4;
    if (i >= total) return;
    int e = (int)(i >> 2), h = (int)(i & 3);
    float mm = m[(size_t)(off_d + dst[e]) * 4 + h];
    if (mm < -1e37f) mm = 0.f;
    float ee = expf(lg[i] - mm);
    lg[i] = ee;
    atomicAdd(&z[(size_t)(off_d + dst[e]) * 4 + h], ee);
}

__global__ void hgt_scatter_kernel(const int* __restrict__ src, const int* __restrict__ dst,
                                   const float* __restrict__ lg, const float* __restrict__ z,
                                   const float* __restrict__ vrel, float* __restrict__ agg,
                                   int E, int fo, int D, int off_d, int relrow) {
    long long i = (long long)blockIdx.x * blockDim.x + threadIdx.x;
    long long total = (long long)E * fo;
    if (i >= total) return;
    int e = (int)(i / fo), f = (int)(i % fo);
    int h = f / D;
    int d = dst[e];
    float a = lg[(size_t)e * 4 + h] / (z[(size_t)(off_d + d) * 4 + h] + 1e-16f);
    atomicAdd(&agg[(size_t)(off_d + d) * fo + f],
              a * vrel[(size_t)(relrow + src[e]) * fo + f]);
}

__global__ void gelu_kernel(const float* __restrict__ x, float* __restrict__ y, long long n) {
    long long i = (long long)blockIdx.x * blockDim.x + threadIdx.x;
    if (i >= n) return;
    float v = x[i];
    const float c = 0.7978845608028654f;
    float t = tanhf(c * (v + 0.044715f * v * v * v));
    y[i] = 0.5f * v * (1.f + t);
}

__global__ void gate_merge_kernel(const float* __restrict__ gb, const float* __restrict__ bgate,
                                  const float* __restrict__ pna, const float* __restrict__ hgt,
                                  float* __restrict__ merged, int N, int F) {
    long long i = (long long)blockIdx.x * blockDim.x + threadIdx.x;
    long long total = (long long)N * F;
    if (i >= total) return;
    int f = (int)(i % F);
    float g = 1.f / (1.f + expf(-(gb[i] + bgate[f])));
    merged[i] = g * pna[i] + (1.f - g) * hgt[i];
}

// ---------------------------------------------------------------------
// Host side
// ---------------------------------------------------------------------
static inline void gemm(hipStream_t s, const float* A, int lda, const float* W, int ldw,
                        const float* bias, float* C, int ldc, int M, int N, int K, int flags) {
    // preconditions: M%32==0, K%32==0, N%4==0 (true for every call below)
    dim3 g((M + 31) / 32, (N + 63) / 64);
    wmma_gemm_kernel<<<g, 256, 0, s>>>(A, lda, W, ldw, bias, C, ldc, M, N, K, flags);
}
static inline unsigned gblocks(long long total) { return (unsigned)((total + 255) / 256); }

extern "C" void kernel_launch(void* const* d_in, const int* in_sizes, int n_in,
                              void* d_out, int out_size, void* d_ws, size_t ws_size,
                              hipStream_t stream) {
    (void)in_sizes; (void)n_in; (void)out_size; (void)ws_size;
    int ii = 0;
    auto NF = [&]() { return (const float*)d_in[ii++]; };
    auto NI = [&]() { return (const int*)d_in[ii++]; };

    // jax pytree flatten order: dict keys sorted at every level, lists in order.
    const float* avg_log = NF();                 // avg_log_deg (scalar)
    const int* E_dg = NI();                      // disease__to__gene [2,64000]
    const int* E_rg = NI();                      // drug__to__gene    [2,64000]
    const int* E_gd = NI();                      // gene__to__disease [2,32000]
    const int* E_gr = NI();                      // gene__to__drug    [2,32000]
    const int* E_gg = NI();                      // gene__to__gene    [2,128000]
    const float* hx_t[3]; hx_t[0] = NF(); hx_t[1] = NF(); hx_t[2] = NF();  // disease,drug,gene
    const int*   homo_ei = NI();                 // [2,320000]
    const float* homo_x  = NF();                 // [40000,128]
    // params (sorted keys): gate, hgt, hgt_norm, out_norm, out_proj, pna, pna_norm
    const float* gate_W = NF(); const float* gate_b = NF();
    struct HgtL { const float *WkW[3], *Wkb[3], *WoW[3], *Wob[3], *WqW[3], *Wqb[3],
                              *WvW[3], *Wvb[3], *arel[5], *mrel[5], *prior[5]; };
    HgtL hgt[2];
    for (int l = 0; l < 2; ++l) {
        for (int t = 0; t < 3; ++t) { hgt[l].WkW[t] = NF(); hgt[l].Wkb[t] = NF(); }
        for (int t = 0; t < 3; ++t) { hgt[l].WoW[t] = NF(); hgt[l].Wob[t] = NF(); }
        for (int t = 0; t < 3; ++t) { hgt[l].WqW[t] = NF(); hgt[l].Wqb[t] = NF(); }
        for (int t = 0; t < 3; ++t) { hgt[l].WvW[t] = NF(); hgt[l].Wvb[t] = NF(); }
        for (int r = 0; r < 5; ++r) hgt[l].arel[r] = NF();
        for (int r = 0; r < 5; ++r) hgt[l].mrel[r] = NF();
        for (int r = 0; r < 5; ++r) hgt[l].prior[r] = NF();
        ii += 3;  // skip gates (dims never match -> unused in eval graph)
    }
    const float *hnb[2][3], *hng[2][3];
    for (int l = 0; l < 2; ++l)
        for (int t = 0; t < 3; ++t) { hnb[l][t] = NF(); hng[l][t] = NF(); }  // {b,g} sorted
    const float* onb = NF(); const float* ong = NF();        // out_norm {b,g}
    const float* opW = NF(); const float* opb = NF();        // out_proj {W,b}
    struct PnaL { const float *Wpi, *Wpj, *bpre, *linW, *linb, *postW, *postb; };
    PnaL pna[2];
    for (int l = 0; l < 2; ++l) {
        pna[l].Wpi = NF(); pna[l].Wpj = NF(); pna[l].bpre = NF();
        pna[l].linW = NF(); pna[l].linb = NF();
        pna[l].postW = NF(); pna[l].postb = NF();
    }
    const float *pnb[2], *png[2];
    for (int l = 0; l < 2; ++l) { pnb[l] = NF(); png[l] = NF(); }

    // ---------------- workspace ----------------
    float* wsf = (float*)d_ws;
    size_t off = 0;
    auto walloc = [&](size_t n) { float* p = wsf + off; off += n; return p; };
    const size_t NB = 40000ull * 256;
    float* cnt  = walloc(40000);
    float* amp  = walloc(40000);
    float* att  = walloc(40000);
    float* bufA = walloc(NB);            // pna_out (final [40000,128])
    float* bufB = walloc(NB);            // pna l0 act / hq / hgt pre-LN
    float* bufC = walloc(NB);            // pre_i / post acc / hk / gate / out_proj res
    float* bufD = walloc(NB);            // pre_j / hv / merged
    float* bufE = walloc(88000ull * 256 * 2);  // aggr (40000x4F) / krel+vrel
    float* bufF = walloc(NB);            // z2 / agg
    float* bufG = walloc(NB);            // z3 / gelu tmp
    float* bufH = walloc(NB);            // hgt layer0 output
    float* bufI = walloc(NB);            // hgt layer1 output
    float* lgbuf = walloc(320000ull * 4);
    float* mbuf  = walloc(160000);
    float* zbuf  = walloc(160000);

    const int* h_src = homo_ei;
    const int* h_dst = homo_ei + 320000;

    // ================= PNA branch =================
    fill_kernel<<<gblocks(40000), 256, 0, stream>>>(cnt, 0.f, 40000);
    count_kernel<<<gblocks(320000), 256, 0, stream>>>(h_dst, 320000, cnt);
    degscale_kernel<<<gblocks(40000), 256, 0, stream>>>(cnt, avg_log, amp, att, 40000);

    const float* x = homo_x;
    for (int l = 0; l < 2; ++l) {
        int fi = (l == 0) ? 128 : 256;
        int fo = (l == 0) ? 256 : 128;
        float* outb = (l == 0) ? bufB : bufA;
        // node-level pre_nn (msg = pre_i[dst] + pre_j[src] + b)
        gemm(stream, x, fi, pna[l].Wpi, fi, nullptr, bufC, fi, 40000, fi, fi, 0);
        gemm(stream, x, fi, pna[l].Wpj, fi, nullptr, bufD, fi, 40000, fi, fi, 0);
        pna_aggr_init_kernel<<<gblocks(40000ll * 4 * fi), 256, 0, stream>>>(bufE, 40000, fi);
        pna_edge_kernel<<<gblocks(320000ll * fi), 256, 0, stream>>>(
            h_src, h_dst, bufC, bufD, pna[l].bpre, bufE, 320000, fi);
        pna_finalize_kernel<<<gblocks(40000ll * fi), 256, 0, stream>>>(bufE, cnt, 40000, fi);
        // post_nn decomposed: out = x@P0 + aggr@P1 + amp*(aggr@P2) + att*(aggr@P3) + b
        const float* Wp = pna[l].postW;   // [(13*fi), fo]
        gemm(stream, x,    fi,     Wp,                        fo, nullptr, bufC, fo, 40000, fo, fi,     0);
        gemm(stream, bufE, 4 * fi, Wp + (size_t)fi * fo,      fo, nullptr, bufC, fo, 40000, fo, 4 * fi, GEMM_ACC);
        gemm(stream, bufE, 4 * fi, Wp + (size_t)5 * fi * fo,  fo, nullptr, bufF, fo, 40000, fo, 4 * fi, 0);
        gemm(stream, bufE, 4 * fi, Wp + (size_t)9 * fi * fo,  fo, nullptr, bufG, fo, 40000, fo, 4 * fi, 0);
        pna_combine_kernel<<<gblocks(40000ll * fo), 256, 0, stream>>>(
            bufC, bufF, bufG, amp, att, pna[l].postb, 40000, fo);
        gemm(stream, bufC, fo, pna[l].linW, fo, pna[l].linb, outb, fo, 40000, fo, fo, GEMM_BIAS);
        // LN (+ReLU on layer0); residual never applies (128->256->128)
        layernorm_kernel<<<40000, 256, 0, stream>>>(outb, png[l], pnb[l], nullptr, outb, fo, l == 0 ? 1 : 0);
        x = outb;
    }
    // pna_out = bufA [40000,128]

    // ================= HGT branch =================
    struct Rel { const int* ei; int E, soff, doff, Ns, relrow, lgoff; };
    const Rel rels[5] = {
        {E_dg,  64000, 0,     16000, 8000,  0,     0},
        {E_rg,  64000, 8000,  16000, 8000,  8000,  64000},
        {E_gd,  32000, 16000, 0,     24000, 16000, 128000},
        {E_gr,  32000, 16000, 8000,  24000, 40000, 160000},
        {E_gg, 128000, 16000, 16000, 24000, 64000, 192000},
    };
    const int Nt[3]  = {8000, 8000, 24000};
    const int offT[3] = {0, 8000, 16000};
    float* krel = bufE;
    float* vrel = bufE + 88000ull * 256;

    for (int l = 0; l < 2; ++l) {
        int fi = (l == 0) ? 128 : 256;
        int fo = (l == 0) ? 256 : 128;
        int D  = fo / 4;
        const float* xt[3];
        for (int t = 0; t < 3; ++t)
            xt[t] = (l == 0) ? hx_t[t] : (bufH + (size_t)offT[t] * 256);
        // q/k/v projections (with bias)
        for (int t = 0; t < 3; ++t) {
            gemm(stream, xt[t], fi, hgt[l].WqW[t], fo, hgt[l].Wqb[t],
                 bufB + (size_t)offT[t] * fo, fo, Nt[t], fo, fi, GEMM_BIAS);
            gemm(stream, xt[t], fi, hgt[l].WkW[t], fo, hgt[l].Wkb[t],
                 bufC + (size_t)offT[t] * fo, fo, Nt[t], fo, fi, GEMM_BIAS);
            gemm(stream, xt[t], fi, hgt[l].WvW[t], fo, hgt[l].Wvb[t],
                 bufD + (size_t)offT[t] * fo, fo, Nt[t], fo, fi, GEMM_BIAS);
        }
        // per-relation per-head transforms: k_rel = k @ a_rel[h], v_rel = v @ m_rel[h]
        for (int r = 0; r < 5; ++r) {
            for (int h = 0; h < 4; ++h) {
                gemm(stream, bufC + (size_t)rels[r].soff * fo + h * D, fo,
                     hgt[l].arel[r] + (size_t)h * D * D, D, nullptr,
                     krel + (size_t)rels[r].relrow * fo + h * D, fo,
                     rels[r].Ns, D, D, 0);
                gemm(stream, bufD + (size_t)rels[r].soff * fo + h * D, fo,
                     hgt[l].mrel[r] + (size_t)h * D * D, D, nullptr,
                     vrel + (size_t)rels[r].relrow * fo + h * D, fo,
                     rels[r].Ns, D, D, 0);
            }
        }
        // attention: shared softmax across all relations targeting a type
        fill_kernel<<<gblocks(160000), 256, 0, stream>>>(mbuf, -3.0e38f, 160000);
        fill_kernel<<<gblocks(160000), 256, 0, stream>>>(zbuf, 0.f, 160000);
        fill_kernel<<<gblocks(40000ll * fo), 256, 0, stream>>>(bufF, 0.f, 40000ll * fo);
        for (int r = 0; r < 5; ++r)
            hgt_logit_kernel<<<gblocks((long long)rels[r].E * 4), 256, 0, stream>>>(
                rels[r].ei, rels[r].ei + rels[r].E, bufB, krel, hgt[l].prior[r],
                lgbuf + (size_t)rels[r].lgoff * 4, mbuf, rels[r].E, fo, D,
                rels[r].doff, rels[r].relrow);
        for (int r = 0; r < 5; ++r)
            hgt_exp_kernel<<<gblocks((long long)rels[r].E * 4), 256, 0, stream>>>(
                rels[r].ei + rels[r].E, lgbuf + (size_t)rels[r].lgoff * 4, mbuf, zbuf,
                rels[r].E, rels[r].doff);
        for (int r = 0; r < 5; ++r)
            hgt_scatter_kernel<<<gblocks((long long)rels[r].E * fo), 256, 0, stream>>>(
                rels[r].ei, rels[r].ei + rels[r].E, lgbuf + (size_t)rels[r].lgoff * 4,
                zbuf, vrel, bufF, rels[r].E, fo, D, rels[r].doff, rels[r].relrow);
        // o = gelu(agg) @ Wo + b ; skip never applies (dims mismatch)
        gelu_kernel<<<gblocks(40000ll * fo), 256, 0, stream>>>(bufF, bufG, 40000ll * fo);
        float* hout = (l == 0) ? bufH : bufI;
        for (int t = 0; t < 3; ++t) {
            gemm(stream, bufG + (size_t)offT[t] * fo, fo, hgt[l].WoW[t], fo, hgt[l].Wob[t],
                 bufB + (size_t)offT[t] * fo, fo, Nt[t], fo, fo, GEMM_BIAS);
            layernorm_kernel<<<Nt[t], 256, 0, stream>>>(
                bufB + (size_t)offT[t] * fo, hng[l][t], hnb[l][t], nullptr,
                hout + (size_t)offT[t] * fo, fo, l == 0 ? 1 : 0);
        }
    }
    // hgt_out = bufI [40000,128] (types contiguous: disease|drug|gene)

    // ================= gated merge =================
    gemm(stream, bufA, 128, gate_W,             128, nullptr, bufC, 128, 40000, 128, 128, 0);
    gemm(stream, bufI, 128, gate_W + 128 * 128, 128, nullptr, bufC, 128, 40000, 128, 128, GEMM_ACC);
    gate_merge_kernel<<<gblocks(40000ll * 128), 256, 0, stream>>>(
        bufC, gate_b, bufA, bufI, bufD, 40000, 128);
    gemm(stream, bufD, 128, opW, 128, opb, bufC, 128, 40000, 128, 128, GEMM_BIAS);
    layernorm_kernel<<<40000, 256, 0, stream>>>(bufC, ong, onb, homo_x, (float*)d_out, 128, 0);
}
